// TemporalAttention_34892314313714
// MI455X (gfx1250) — matrix operand order repaired and checked by reading, not compiled, not run on hardware
//
#include <hip/hip_runtime.h>
#include <hip/hip_bf16.h>

// ---------------------------------------------------------------------------
// TemporalAttention for MI455X (gfx1250, wave32, WMMA bf16 16x16x32).
// B=4, T=8, C=128, H=W=96 (HW=9216), heads=4, head_dim=32.
// Only the last query timestep is used by the reference -> q computed only
// for t=T-1. Intermediates stored bf16 to halve HBM traffic (chip is
// bandwidth-bound here: ~0.66 GB moved vs ~22 GFLOP of bf16 WMMA work).
// ---------------------------------------------------------------------------

typedef __attribute__((ext_vector_type(16))) __bf16 v16bf;
typedef __attribute__((ext_vector_type(8)))  __bf16 v8bf;
typedef __attribute__((ext_vector_type(8)))  float  v8f;

#define B_      4
#define T_      8
#define C_      128
#define HW_     9216
#define NH_     4
#define HD_     32
#define NSLICE  (B_ * T_)          // 32
#define SLICE_N (C_ * HW_)         // 1179648
#define KPAD    8                  // bf16 pad -> row stride 272B (16B aligned)

// ---- workspace layout (bytes) ---------------------------------------------
#define OFF_STATS 0u                                   // 32 * {sum, sumsq}
#define OFF_MR    256u                                 // 32 * {mean, rstd}
#define OFF_WQKV  512u                                 // 384*128 bf16
#define OFF_WPROJ (OFF_WQKV + 3u*C_*C_*2u)             // 128*128 bf16
#define OFF_K     ((OFF_WPROJ + (unsigned)(C_*C_*2) + 255u) & ~255u)
#define SZ_KV     ((size_t)NSLICE * C_ * HW_ * 2)      // 75,497,472 B each
#define OFF_V     (OFF_K + SZ_KV)
#define OFF_Q     (OFF_V + SZ_KV)
#define SZ_Q      ((size_t)B_ * C_ * HW_ * 2)          // 9,437,184 B
#define OFF_AO    (OFF_Q + SZ_Q)
// total ws usage: ~170 MB

// ---------------------------------------------------------------------------
// Pass 1a: per-(b,t) partial sum / sum-of-squares of (x + pos_enc)
// ---------------------------------------------------------------------------
__global__ __launch_bounds__(256)
void k_stats_partial(const float* __restrict__ x, const float* __restrict__ pos,
                     float* __restrict__ part) {
  const int s = blockIdx.y;             // (b*T + t)
  const int t = s % T_;
  const float* xs = x + (size_t)s * SLICE_N;
  const int base = blockIdx.x * 4096;   // 288 blocks/slice * 4096 elems
  float sum = 0.f, sq = 0.f;
#pragma unroll
  for (int j = 0; j < 4; ++j) {
    const int e = base + j * 1024 + threadIdx.x * 4;   // float4 never crosses a C row
    const int c = e / HW_;
    const float p = pos[t * C_ + c];
    const float4 v = *(const float4*)(xs + e);
    const float a0 = v.x + p, a1 = v.y + p, a2 = v.z + p, a3 = v.w + p;
    sum += a0 + a1 + a2 + a3;
    sq  += a0 * a0 + a1 * a1 + a2 * a2 + a3 * a3;
  }
  __shared__ float ls[256], lq[256];
  ls[threadIdx.x] = sum; lq[threadIdx.x] = sq;
  __syncthreads();
  for (int off = 128; off > 0; off >>= 1) {
    if ((int)threadIdx.x < off) {
      ls[threadIdx.x] += ls[threadIdx.x + off];
      lq[threadIdx.x] += lq[threadIdx.x + off];
    }
    __syncthreads();
  }
  if (threadIdx.x == 0) {
    atomicAdd(&part[s * 2 + 0], ls[0]);
    atomicAdd(&part[s * 2 + 1], lq[0]);
  }
}

// ---------------------------------------------------------------------------
// Pass 1b: finalize mean / rstd
// ---------------------------------------------------------------------------
__global__ void k_stats_final(const float* __restrict__ part, float* __restrict__ mr) {
  const int s = threadIdx.x;
  if (s < NSLICE) {
    const float n = (float)SLICE_N;
    const float mean = part[s * 2] / n;
    const float var  = part[s * 2 + 1] / n - mean * mean;
    mr[s * 2]     = mean;
    mr[s * 2 + 1] = rsqrtf(var + 1e-5f);
  }
}

// ---------------------------------------------------------------------------
// Weight conversion fp32 -> bf16 (once per call; tiny)
// ---------------------------------------------------------------------------
__global__ void k_cvt_w(const float* __restrict__ qkvw, const float* __restrict__ projw,
                        __bf16* __restrict__ wq, __bf16* __restrict__ wp) {
  const int i = blockIdx.x * 256 + threadIdx.x;
  if (i < 3 * C_ * C_) wq[i] = (__bf16)qkvw[i];
  if (i < C_ * C_)     wp[i] = (__bf16)projw[i];
}

// ---------------------------------------------------------------------------
// WMMA fragment helpers (CDNA5 wave32 layouts, CDNA5 ISA §7.12.2)
//  A 16x32 bf16: lane l<16 -> row l, K in {0..7, 16..23}; lane>=16 -> K+8
//  B 32x16 bf16: lane l<16 -> col l, K in {0..7, 16..23}; lane>=16 -> K+8
//  C/D 16x16 f32: lane l -> N=l&15; VGPR r -> M = r + (l>=16 ? 8 : 0)
// ---------------------------------------------------------------------------
__device__ __forceinline__ v16bf frag_a(const __bf16* row, int kb, bool hi) {
  const v8bf lo = *(const v8bf*)(row + kb + (hi ? 8 : 0));
  const v8bf hi8 = *(const v8bf*)(row + kb + 16 + (hi ? 8 : 0));
  return __builtin_shufflevector(lo, hi8, 0,1,2,3,4,5,6,7,8,9,10,11,12,13,14,15);
}
__device__ __forceinline__ v16bf frag_b(const __bf16* col, int kb, bool hi) {
  const v8bf lo = *(const v8bf*)(col + kb + (hi ? 8 : 0));
  const v8bf hi8 = *(const v8bf*)(col + kb + 16 + (hi ? 8 : 0));
  return __builtin_shufflevector(lo, hi8, 0,1,2,3,4,5,6,7,8,9,10,11,12,13,14,15);
}

// ---------------------------------------------------------------------------
// Pass 2: fused normalize + QKV GEMM.  Per (b,t): D[384x9216] = Wqkv @ xn.
// Block = 128 threads (4 waves), computes 64(M) x 64(N); K=128 fully staged.
// Each wave: 16(M) x 64(N) via 4 k-steps x 4 n-subtiles of v_wmma bf16.
// M tiles of q (rows < 128) skipped unless t == T-1.
// ---------------------------------------------------------------------------
__global__ __launch_bounds__(128)
void k_qkv_gemm(const float* __restrict__ x, const float* __restrict__ pos,
                const float* __restrict__ nw, const float* __restrict__ nb,
                const __bf16* __restrict__ wq, const float* __restrict__ mr,
                __bf16* __restrict__ kws, __bf16* __restrict__ vws,
                __bf16* __restrict__ qws) {
  const int s = blockIdx.z;                 // b*T + t
  const int b = s / T_, t = s % T_;
  const int mbase = blockIdx.y * 64;        // 0..320
  if (t != T_ - 1 && mbase < C_) return;    // unused q rows
  const int nbase = blockIdx.x * 64;

  __shared__ __bf16 Al[64][C_ + KPAD];      // weights  [m][k]
  __shared__ __bf16 Bl[64][C_ + KPAD];      // activ.   [n][k] (transposed)

  const int tid = threadIdx.x;
  const float mean = mr[s * 2], rstd = mr[s * 2 + 1];

  { // stage A: 64 rows x 128 K, 16B vector loads
    const __bf16* src = wq + (size_t)mbase * C_;
    for (int i = tid; i < 64 * 16; i += 128) {
      const int r = i >> 4, cb = (i & 15) * 8;
      *(uint4*)&Al[r][cb] = *(const uint4*)&src[r * C_ + cb];
    }
  }
  { // stage B: read x coalesced along HW, normalize, store transposed
    const float* xs = x + (size_t)s * SLICE_N + nbase;
    for (int i = tid; i < 64 * C_; i += 128) {
      const int c = i >> 6, n = i & 63;
      const float v = xs[(size_t)c * HW_ + n] + pos[t * C_ + c];
      Bl[n][c] = (__bf16)((v - mean) * rstd * nw[c] + nb[c]);
    }
  }
  __syncthreads();

  const int wave = tid >> 5, lane = tid & 31;
  const int l16 = lane & 15;
  const bool hi = lane >= 16;

  v8f acc[4] = {{}, {}, {}, {}};
#pragma unroll
  for (int kk = 0; kk < 4; ++kk) {
    const int kb = kk * 32;
    const v16bf a = frag_a(&Al[wave * 16 + l16][0], kb, hi);
#pragma unroll
    for (int nt = 0; nt < 4; ++nt) {
      const v16bf bf = frag_b(&Bl[nt * 16 + l16][0], kb, hi);
      acc[nt] = __builtin_amdgcn_wmma_f32_16x16x32_bf16(
          false, a, false, bf, (short)0, acc[nt], false, false);
    }
  }

  // store bf16, routed by row block (uniform per block: 64 | 128-boundaries)
#pragma unroll
  for (int nt = 0; nt < 4; ++nt) {
    const int n = nbase + nt * 16 + l16;
#pragma unroll
    for (int r = 0; r < 8; ++r) {
      const int m = mbase + wave * 16 + r + (hi ? 8 : 0);
      const __bf16 bv = (__bf16)acc[nt][r];
      if (m < C_)                    // q (t == T-1 only)
        qws[((size_t)b * C_ + m) * HW_ + n] = bv;
      else if (m < 2 * C_)           // k
        kws[((size_t)s * C_ + (m - C_)) * HW_ + n] = bv;
      else                           // v
        vws[((size_t)s * C_ + (m - 2 * C_)) * HW_ + n] = bv;
    }
  }
}

// ---------------------------------------------------------------------------
// Pass 3: attention, last query row only. One lane per (b, head, pixel);
// all global accesses coalesced along HW.
// ---------------------------------------------------------------------------
__global__ __launch_bounds__(256)
void k_attn(const __bf16* __restrict__ qws, const __bf16* __restrict__ kws,
            const __bf16* __restrict__ vws, __bf16* __restrict__ ao) {
  const int gid = blockIdx.x * 256 + threadIdx.x;
  if (gid >= B_ * NH_ * HW_) return;
  const int hw = gid % HW_;
  const int bn = gid / HW_;
  const int h  = bn % NH_;
  const int b  = bn / NH_;
  const float scale = 0.17677669529663688f;  // 32^-0.5

  float q[HD_];
#pragma unroll
  for (int d = 0; d < HD_; ++d)
    q[d] = (float)qws[((size_t)b * C_ + h * HD_ + d) * HW_ + hw];

  float logit[T_];
#pragma unroll
  for (int t = 0; t < T_; ++t) {
    const __bf16* kp = kws + ((size_t)(b * T_ + t) * C_ + h * HD_) * HW_ + hw;
    float acc = 0.f;
#pragma unroll
    for (int d = 0; d < HD_; ++d) acc += q[d] * (float)kp[(size_t)d * HW_];
    logit[t] = acc * scale;
  }
  float mx = logit[0];
#pragma unroll
  for (int t = 1; t < T_; ++t) mx = fmaxf(mx, logit[t]);
  float p[T_], den = 0.f;
#pragma unroll
  for (int t = 0; t < T_; ++t) { p[t] = __expf(logit[t] - mx); den += p[t]; }
  const float inv = 1.f / den;

  float o[HD_];
#pragma unroll
  for (int d = 0; d < HD_; ++d) o[d] = 0.f;
#pragma unroll
  for (int t = 0; t < T_; ++t) {
    const __bf16* vp = vws + ((size_t)(b * T_ + t) * C_ + h * HD_) * HW_ + hw;
    const float w = p[t] * inv;
#pragma unroll
    for (int d = 0; d < HD_; ++d) o[d] += w * (float)vp[(size_t)d * HW_];
  }
#pragma unroll
  for (int d = 0; d < HD_; ++d)
    ao[((size_t)b * C_ + h * HD_ + d) * HW_ + hw] = (__bf16)o[d];
}

// ---------------------------------------------------------------------------
// Pass 4: output projection, fp32 result. Per b: Y[128x9216] = Wproj @ ao.
// ---------------------------------------------------------------------------
__global__ __launch_bounds__(128)
void k_proj_gemm(const __bf16* __restrict__ wp, const __bf16* __restrict__ ao,
                 float* __restrict__ out) {
  const int b = blockIdx.z;
  const int mbase = blockIdx.y * 64;
  const int nbase = blockIdx.x * 64;

  __shared__ __bf16 Al[64][C_ + KPAD];
  __shared__ __bf16 Bl[64][C_ + KPAD];
  const int tid = threadIdx.x;

  {
    const __bf16* src = wp + (size_t)mbase * C_;
    for (int i = tid; i < 64 * 16; i += 128) {
      const int r = i >> 4, cb = (i & 15) * 8;
      *(uint4*)&Al[r][cb] = *(const uint4*)&src[r * C_ + cb];
    }
  }
  {
    const __bf16* src = ao + (size_t)b * C_ * HW_ + nbase;
    for (int i = tid; i < 64 * C_; i += 128) {
      const int c = i >> 6, n = i & 63;
      Bl[n][c] = src[(size_t)c * HW_ + n];
    }
  }
  __syncthreads();

  const int wave = tid >> 5, lane = tid & 31;
  const int l16 = lane & 15;
  const bool hi = lane >= 16;

  v8f acc[4] = {{}, {}, {}, {}};
#pragma unroll
  for (int kk = 0; kk < 4; ++kk) {
    const int kb = kk * 32;
    const v16bf a = frag_a(&Al[wave * 16 + l16][0], kb, hi);
#pragma unroll
    for (int nt = 0; nt < 4; ++nt) {
      const v16bf bf = frag_b(&Bl[nt * 16 + l16][0], kb, hi);
      acc[nt] = __builtin_amdgcn_wmma_f32_16x16x32_bf16(
          false, a, false, bf, (short)0, acc[nt], false, false);
    }
  }
#pragma unroll
  for (int nt = 0; nt < 4; ++nt) {
    const int n = nbase + nt * 16 + l16;
#pragma unroll
    for (int r = 0; r < 8; ++r) {
      const int m = mbase + wave * 16 + r + (hi ? 8 : 0);
      out[((size_t)b * C_ + m) * HW_ + n] = acc[nt][r];
    }
  }
}

// ---------------------------------------------------------------------------
extern "C" void kernel_launch(void* const* d_in, const int* in_sizes, int n_in,
                              void* d_out, int out_size, void* d_ws, size_t ws_size,
                              hipStream_t stream) {
  const float* x     = (const float*)d_in[0];
  const float* pos   = (const float*)d_in[1];
  const float* nw    = (const float*)d_in[2];
  const float* nb    = (const float*)d_in[3];
  const float* qkvw  = (const float*)d_in[4];
  const float* projw = (const float*)d_in[5];
  float* out = (float*)d_out;

  char* ws = (char*)d_ws;
  float*  part = (float*)(ws + OFF_STATS);
  float*  mr   = (float*)(ws + OFF_MR);
  __bf16* wq   = (__bf16*)(ws + OFF_WQKV);
  __bf16* wp   = (__bf16*)(ws + OFF_WPROJ);
  __bf16* kws  = (__bf16*)(ws + OFF_K);
  __bf16* vws  = (__bf16*)(ws + OFF_V);
  __bf16* qws  = (__bf16*)(ws + OFF_Q);
  __bf16* ao   = (__bf16*)(ws + OFF_AO);

  hipMemsetAsync(part, 0, NSLICE * 2 * sizeof(float), stream);

  k_cvt_w<<<(3 * C_ * C_ + 255) / 256, 256, 0, stream>>>(qkvw, projw, wq, wp);
  k_stats_partial<<<dim3(SLICE_N / 4096, NSLICE), 256, 0, stream>>>(x, pos, part);
  k_stats_final<<<1, 32, 0, stream>>>(part, mr);
  k_qkv_gemm<<<dim3(HW_ / 64, 6, NSLICE), 128, 0, stream>>>(
      x, pos, nw, nb, wq, mr, kws, vws, qws);
  k_attn<<<(B_ * NH_ * HW_ + 255) / 256, 256, 0, stream>>>(qws, kws, vws, ao);
  k_proj_gemm<<<dim3(HW_ / 64, 2, B_), 128, 0, stream>>>(wp, ao, out);
}
